// Corr_recontruction_71390946394404
// MI455X (gfx1250) — matrix operand order repaired and checked by reading, not compiled, not run on hardware
//
#include <hip/hip_runtime.h>

// ---------------- problem constants ----------------
#define B_   2
#define CF   32
#define H_   256
#define W_   256
#define RR   6
#define DWIN 13          // window side
#define NDISP (DWIN*DWIN)
#define HP   268         // padded height (H + 2R)
#define WP   272         // padded width stride (268 rounded up so tile-1 column overreads stay in-bounds)
#define CSTR 172         // corr row stride (16B-aligned, holds 169 + 3 pad slots)

typedef __attribute__((ext_vector_type(16))) __bf16 v16bf;
typedef __attribute__((ext_vector_type(8)))  float  v8f;

__device__ __forceinline__ unsigned short bf_hi(float v) {
    __bf16 b = (__bf16)v;
    return __builtin_bit_cast(unsigned short, b);
}
__device__ __forceinline__ unsigned short bf_lo(float v) {
    __bf16 b = (__bf16)v;
    float res = v - (float)b;
    __bf16 l = (__bf16)res;
    return __builtin_bit_cast(unsigned short, l);
}

// ---------------- prep kernels ----------------

__global__ void zero_fill(float4* __restrict__ dst, size_t n4) {
    size_t i = (size_t)blockIdx.x * blockDim.x + threadIdx.x;
    if (i < n4) dst[i] = make_float4(0.f, 0.f, 0.f, 0.f);
}

// feats_t [b,c,h,w] f32 -> Tt_hi/Tt_lo [b,y,x,c] bf16 (channel-innermost, hi/lo split)
__global__ __launch_bounds__(256) void conv_t(const float* __restrict__ src,
                                              unsigned short* __restrict__ dh,
                                              unsigned short* __restrict__ dl) {
    __shared__ float tile[CF][33];
    const int tx = threadIdx.x & 31;
    const int ty = threadIdx.x >> 5;
    const int blk = blockIdx.x;                 // b*h*(w/32)
    const int xt = blk & 7;
    const int y  = (blk >> 3) & (H_ - 1);
    const int bb = blk >> 11;
    const int x0 = xt << 5;
#pragma unroll
    for (int cc = 0; cc < 4; ++cc) {
        int c = ty + 8 * cc;
        tile[c][tx] = src[(((size_t)bb * CF + c) * H_ + y) * W_ + x0 + tx];
    }
    __syncthreads();
#pragma unroll
    for (int xx = 0; xx < 4; ++xx) {
        int xl = ty + 8 * xx;
        float v = tile[tx][xl];                 // channel = tx
        size_t o = ((size_t)(bb * H_ + y) * W_ + x0 + xl) * CF + tx;
        dh[o] = bf_hi(v);
        dl[o] = bf_lo(v);
    }
}

// feats_r -> zero-padded Rp_hi/Rp_lo [b, y+R, x+R, c]
__global__ __launch_bounds__(256) void conv_r(const float* __restrict__ src,
                                              unsigned short* __restrict__ dh,
                                              unsigned short* __restrict__ dl) {
    __shared__ float tile[CF][33];
    const int tx = threadIdx.x & 31;
    const int ty = threadIdx.x >> 5;
    const int blk = blockIdx.x;
    const int xt = blk & 7;
    const int y  = (blk >> 3) & (H_ - 1);
    const int bb = blk >> 11;
    const int x0 = xt << 5;
#pragma unroll
    for (int cc = 0; cc < 4; ++cc) {
        int c = ty + 8 * cc;
        tile[c][tx] = src[(((size_t)bb * CF + c) * H_ + y) * W_ + x0 + tx];
    }
    __syncthreads();
#pragma unroll
    for (int xx = 0; xx < 4; ++xx) {
        int xl = ty + 8 * xx;
        float v = tile[tx][xl];
        size_t o = ((size_t)(bb * HP + y + RR) * WP + (x0 + xl + RR)) * CF + tx;
        dh[o] = bf_hi(v);
        dl[o] = bf_lo(v);
    }
}

// img_r [b,1,h,w] f32 -> Ip [b, HP, WP] padded f32 (interior only; halo pre-zeroed)
__global__ void conv_img(const float* __restrict__ src, float* __restrict__ dst) {
    int i = blockIdx.x * 256 + threadIdx.x;
    if (i >= B_ * H_ * W_) return;
    int x = i & (W_ - 1);
    int y = (i >> 8) & (H_ - 1);
    int bb = i >> 16;
    dst[(size_t)(bb * HP + y + RR) * WP + (x + RR)] = src[i];
}

// ---------------- main kernel ----------------
// 128 threads = 4 waves/block; each wave owns one 16-pixel x-strip.
__global__ __launch_bounds__(128) void corr_main(
    const unsigned short* __restrict__ Tt_hi, const unsigned short* __restrict__ Tt_lo,
    const unsigned short* __restrict__ Rp_hi, const unsigned short* __restrict__ Rp_lo,
    const float* __restrict__ Ip, float* __restrict__ out)
{
    // per-wave: 16 pixel rows of stride CSTR + 32 dump slots
    __shared__ float lds_corr[4][16 * CSTR + 32];
    __shared__ float lds_img[4][DWIN * 32];     // [wave][dy][n], n = 0..31 (28..31 zeroed)

    const int wave = threadIdx.x >> 5;
    const int lane = threadIdx.x & 31;
    const int strip = blockIdx.x * 4 + wave;    // b * h * (w/16) = 8192 strips
    const int xseg = strip & 15;
    const int y    = (strip >> 4) & (H_ - 1);
    const int bb   = strip >> 12;
    const int x0   = xseg << 4;

    float* corrB = lds_corr[wave];
    float* imgB  = lds_img[wave];

    const int g = lane >> 4;                    // lane half
    const int m = lane & 15;

    // ---- stage padded img rows y..y+12, cols x0..x0+27 into LDS; zero cols 28..31 ----
    const float* ipBase = Ip + (size_t)bb * HP * WP;
    for (int i = lane; i < DWIN * 32; i += 32) {
        int dy = i >> 5, n = i & 31;
        imgB[i] = (n < 28) ? ipBase[(size_t)(y + dy) * WP + (x0 + n)] : 0.f;
    }
    // ---- init corr pad slots (d = 169..171) so masked exp() can never be Inf ----
    for (int i = lane; i < 16 * 3; i += 32) {
        int px = i / 3, sl = 169 + (i - px * 3);
        corrB[px * CSTR + sl] = -1.0e30f;
    }

    // ---- A fragments (feats_t strip); ISA 16-bit A layout:
    //      lane = m + 16g, elem e -> K = 16*(e/8) + 8*g + (e%8)
    union Frag { v16bf v; uint4 q[2]; };
    Frag a_hi, a_lo;
    {
        size_t base = ((size_t)(bb * H_ + y) * W_ + (x0 + m)) * CF;
        const unsigned short* ph = Tt_hi + base;
        const unsigned short* pl = Tt_lo + base;
        const int c0 = 8 * g, c1 = 16 + 8 * g;
        a_hi.q[0] = *(const uint4*)(ph + c0);
        a_hi.q[1] = *(const uint4*)(ph + c1);
        a_lo.q[0] = *(const uint4*)(pl + c0);
        a_lo.q[1] = *(const uint4*)(pl + c1);
    }

    const int dumpIdx = 16 * CSTR + lane;       // per-lane trash slot for invalid band entries

    // ---- per dy: banded GEMM C[m][n] = sum_c T[m,c] * Rpad[y+dy, x0+n, c] ----
    for (int dy = 0; dy < DWIN; ++dy) {
        const size_t rowBase = ((size_t)(bb * HP + (y + dy)) * WP) * CF;
        // prefetch next reference row (gfx1250 global_prefetch_b8)
        if (dy + 1 < DWIN) {
            size_t nb = rowBase + (size_t)WP * CF + (size_t)(x0 + m) * CF;
            __builtin_prefetch(Rp_hi + nb, 0, 1);
            __builtin_prefetch(Rp_lo + nb, 0, 1);
        }
#pragma unroll
        for (int t = 0; t < 2; ++t) {
            const int n = m + 16 * t;           // B column; B layout: lane = n%16 + 16g, elem e -> K = 16*(e/8) + 8g + (e%8)
            const size_t pbase = rowBase + (size_t)(x0 + n) * CF + 8 * g;
            Frag b_hi, b_lo;
            b_hi.q[0] = *(const uint4*)(Rp_hi + pbase);
            b_hi.q[1] = *(const uint4*)(Rp_hi + pbase + 16);
            b_lo.q[0] = *(const uint4*)(Rp_lo + pbase);
            b_lo.q[1] = *(const uint4*)(Rp_lo + pbase + 16);

            // near-fp32 product via bf16 hi/lo split: hi*hi + hi*lo + lo*hi
            v8f c = {};
            c = __builtin_amdgcn_wmma_f32_16x16x32_bf16(false, a_lo.v, false, b_hi.v, (short)0, c, false, false);
            c = __builtin_amdgcn_wmma_f32_16x16x32_bf16(false, a_hi.v, false, b_lo.v, (short)0, c, false, false);
            c = __builtin_amdgcn_wmma_f32_16x16x32_bf16(false, a_hi.v, false, b_hi.v, (short)0, c, false, false);

            // branchless band scatter: C elem r -> M = r + 8g, N = n; dx = N - M in [0,12]
#pragma unroll
            for (int r = 0; r < 8; ++r) {
                const int M = r + 8 * g;
                const int dx = n - M;
                const int idx = ((unsigned)dx <= 12u) ? (M * CSTR + dy * DWIN + dx) : dumpIdx;
                corrB[idx] = c[r];
            }
        }
    }
    __syncthreads();   // order LDS writes before reads (uniform control flow)

    // ---- single-pass softmax over 169 displacements + weighted image sum ----
    // Stabilized by the center-displacement logit (exact softmax is shift-invariant;
    // |logit| for 32-dim N(0,1) dots is << 88, so no overflow without a true max).
    const int p = m;
    const float c0v = corrB[p * CSTR + 84];     // dy=6, dx=6 (zero displacement)

    // lane pair (p, p+16): g=0 handles dx 0..6, g=1 handles dx 7..12 (7th slot masked)
    const float msk6 = (g == 0) ? 1.f : 0.f;
    const float* cbase = corrB + p * CSTR + 7 * g;
    const float* ibase = imgB + p + 7 * g;

    float s = 0.f, wsum = 0.f;
    for (int dy = 0; dy < DWIN; ++dy) {
        const float* cptr = cbase + dy * DWIN;
        const float* iptr = ibase + dy * 32;
#pragma unroll
        for (int j = 0; j < 7; ++j) {
            float e = __expf(cptr[j] - c0v);
            if (j == 6) e *= msk6;              // mask lane-half overhang (never Inf: pad slots = -1e30)
            s    += e;
            wsum += e * iptr[j];
        }
    }
    s    += __shfl_xor(s, 16, 32);
    wsum += __shfl_xor(wsum, 16, 32);

    if (g == 0)
        out[((size_t)bb * H_ + y) * W_ + (x0 + p)] = wsum / s;
}

// ---------------- launch ----------------
extern "C" void kernel_launch(void* const* d_in, const int* in_sizes, int n_in,
                              void* d_out, int out_size, void* d_ws, size_t ws_size,
                              hipStream_t stream) {
    const float* feats_t = (const float*)d_in[0];
    const float* feats_r = (const float*)d_in[1];
    const float* img_r   = (const float*)d_in[2];
    float* out = (float*)d_out;

    char* ws = (char*)d_ws;
    const size_t szT = (size_t)B_ * H_ * W_ * CF * 2;   // 8,388,608 B per hi/lo array
    const size_t szR = (size_t)B_ * HP * WP * CF * 2;   // 9,330,688 B per hi/lo array
    const size_t szI = (size_t)B_ * HP * WP * 4;        //   583,168 B
    // total workspace ~36 MB
    unsigned short* Tt_hi = (unsigned short*)(ws);
    unsigned short* Tt_lo = (unsigned short*)(ws + szT);
    unsigned short* Rp_hi = (unsigned short*)(ws + 2 * szT);
    unsigned short* Rp_lo = (unsigned short*)(ws + 2 * szT + szR);
    float*          Ip    = (float*)         (ws + 2 * szT + 2 * szR);

    // zero-fill padded reference + img buffers (halo)
    const size_t zbytes = 2 * szR + szI;
    zero_fill<<<(unsigned)((zbytes / 16 + 255) / 256), 256, 0, stream>>>(
        (float4*)(ws + 2 * szT), zbytes / 16);

    conv_t  <<<B_ * H_ * (W_ / 32), 256, 0, stream>>>(feats_t, Tt_hi, Tt_lo);
    conv_r  <<<B_ * H_ * (W_ / 32), 256, 0, stream>>>(feats_r, Rp_hi, Rp_lo);
    conv_img<<<(B_ * H_ * W_ + 255) / 256, 256, 0, stream>>>(img_r, Ip);

    // 8192 strips, 4 waves per 128-thread block
    corr_main<<<2048, 128, 0, stream>>>(Tt_hi, Tt_lo, Rp_hi, Rp_lo, Ip, out);
}
